// MambaEncoder_35768487641361
// MI455X (gfx1250) — compile-verified, matrix operand order checked by reading, not run on hardware
//
#include <hip/hip_runtime.h>
#include <hip/hip_bf16.h>
#include <math.h>

// ---- problem dims (from reference) ----
#define L_SEQ   4096
#define DMODEL  256
#define DINNER  512
#define DSTATE  16
#define DTRANK  16
#define KCONV   4
#define NLAYERS 8
#define INCH    224
#define LATD    192
#define XDBP    64          // padded x_dbl row stride (dt 0..15 | B 16..31 | C 32..47 | pad)

typedef __attribute__((ext_vector_type(16))) __bf16 v16bf;
typedef __attribute__((ext_vector_type(8)))  __bf16 v8bf;
typedef __attribute__((ext_vector_type(8)))  float  v8f;

__device__ __forceinline__ float sigmoidf_(float x) { return 1.f / (1.f + __expf(-x)); }

__device__ __forceinline__ v8bf cvt8(float4 a, float4 b) {
    v8bf r;
    r[0] = (__bf16)a.x; r[1] = (__bf16)a.y; r[2] = (__bf16)a.z; r[3] = (__bf16)a.w;
    r[4] = (__bf16)b.x; r[5] = (__bf16)b.y; r[6] = (__bf16)b.z; r[7] = (__bf16)b.w;
    return r;
}

// ---------------------------------------------------------------------------
// Tiled bf16 WMMA GEMM (all dims pre-padded: M%64==0, N%64==0, K%64==0):
//   C[m,n] = act( sum_k A[m,k] * W[n,k] + bias[n] ) + addsrc[m,n]
//   W is [N,K] row-major fp32 (reference layout; rows of W == B columns).
//   TC: store transposed (C[n*ldc + m]) for the final NCHW output.
// Block: 256 threads = 8 wave32s. C tile 64x64; each wave owns a 16x32 strip
// (2 accumulators, A fragment reused), K unrolled x2 -> 4 WMMAs per barrier.
// Staging: global_load_b128 -> v_cvt_pk_bf16_f32 -> ds_store_b128, unguarded.
// ---------------------------------------------------------------------------
template<int ACT, bool TC>
__global__ __launch_bounds__(256)
void gemm_bf16_wmma(const float* __restrict__ A, const float* __restrict__ W,
                    const float* __restrict__ bias, const float* __restrict__ addsrc,
                    float* __restrict__ C, int M, int N, int K,
                    int lda, int ldw, int ldc)
{
    constexpr int BM = 64, BN = 64, BK = 64, PAD = 8;  // row stride 72*2 = 144B (16B multiple)
    __shared__ __bf16 As[BM][BK + PAD];
    __shared__ __bf16 Ws[BN][BK + PAD];

    const int tid  = threadIdx.x;
    const int lane = tid & 31;
    const int wave = tid >> 5;
    const int wm   = wave & 3;          // 4 row tiles of 16
    const int wc   = wave >> 2;         // 2 col strips of 32
    const int m0   = blockIdx.y * BM;
    const int n0   = blockIdx.x * BN;
    const int half = lane >> 4;         // K-half select per ISA fragment layout
    const int r16  = lane & 15;

    const int r    = tid >> 2;          // staging row 0..63
    const int cseg = (tid & 3) * 16;    // staging col segment 0,16,32,48

    v8f acc0 = {}, acc1 = {};

    for (int k0 = 0; k0 < K; k0 += BK) {
        {
            const float* ap = A + (size_t)(m0 + r) * lda + k0 + cseg;
            float4 f0 = *(const float4*)(ap + 0);
            float4 f1 = *(const float4*)(ap + 4);
            float4 f2 = *(const float4*)(ap + 8);
            float4 f3 = *(const float4*)(ap + 12);
            *(v8bf*)&As[r][cseg]     = cvt8(f0, f1);
            *(v8bf*)&As[r][cseg + 8] = cvt8(f2, f3);
        }
        {
            const float* wp = W + (size_t)(n0 + r) * ldw + k0 + cseg;
            float4 f0 = *(const float4*)(wp + 0);
            float4 f1 = *(const float4*)(wp + 4);
            float4 f2 = *(const float4*)(wp + 8);
            float4 f3 = *(const float4*)(wp + 12);
            *(v8bf*)&Ws[r][cseg]     = cvt8(f0, f1);
            *(v8bf*)&Ws[r][cseg + 8] = cvt8(f2, f3);
        }
        __syncthreads();

        // speculative prefetch of next K tile (global_prefetch_b8)
        if (k0 + BK < K) {
            __builtin_prefetch(&A[(size_t)(m0 + r) * lda + k0 + BK + cseg], 0, 1);
            __builtin_prefetch(&W[(size_t)(n0 + r) * ldw + k0 + BK + cseg], 0, 1);
        }

        const int ar  = wm * 16 + r16;
        const int bn0 = wc * 32 + r16;
        const int bn1 = bn0 + 16;
        #pragma unroll
        for (int kk = 0; kk < BK; kk += 32) {
            v8bf a0 = *(const v8bf*)&As[ar][kk + half * 8];
            v8bf a1 = *(const v8bf*)&As[ar][kk + 16 + half * 8];
            v16bf af = __builtin_shufflevector(a0, a1, 0,1,2,3,4,5,6,7,8,9,10,11,12,13,14,15);
            v8bf b0 = *(const v8bf*)&Ws[bn0][kk + half * 8];
            v8bf b1 = *(const v8bf*)&Ws[bn0][kk + 16 + half * 8];
            v16bf bf0 = __builtin_shufflevector(b0, b1, 0,1,2,3,4,5,6,7,8,9,10,11,12,13,14,15);
            v8bf b2 = *(const v8bf*)&Ws[bn1][kk + half * 8];
            v8bf b3 = *(const v8bf*)&Ws[bn1][kk + 16 + half * 8];
            v16bf bf1 = __builtin_shufflevector(b2, b3, 0,1,2,3,4,5,6,7,8,9,10,11,12,13,14,15);

            acc0 = __builtin_amdgcn_wmma_f32_16x16x32_bf16(
                       false, af, false, bf0, (short)0, acc0, false, false);
            acc1 = __builtin_amdgcn_wmma_f32_16x16x32_bf16(
                       false, af, false, bf1, (short)0, acc1, false, false);
        }
        __syncthreads();
    }

    // ---- store: lane n = lane&15, VGPR v -> m = v + 8*(lane>>4) ----
    const int gn0 = n0 + wc * 32 + r16;
    const int gn1 = gn0 + 16;
    #pragma unroll
    for (int v = 0; v < 8; ++v) {
        int gm = m0 + wm * 16 + half * 8 + v;
        #pragma unroll
        for (int t = 0; t < 2; ++t) {
            int gn = t ? gn1 : gn0;
            float rr = t ? acc1[v] : acc0[v];
            if (bias) rr += bias[gn];
            if (ACT == 1) rr = fmaxf(rr, 0.f);                              // relu
            else if (ACT == 2) rr = (rr > 20.f) ? rr : log1pf(__expf(rr));  // softplus
            if (addsrc) rr += addsrc[(size_t)gm * ldc + gn];                // residual
            size_t ci = TC ? (size_t)gn * ldc + gm : (size_t)gm * ldc + gn;
            C[ci] = rr;
        }
    }
}

// ---------------------------------------------------------------------------
// Weight zero-padding: (nl, ns, ks) -> (nl, np, kp)
// ---------------------------------------------------------------------------
__global__ void padw_k(const float* __restrict__ src, float* __restrict__ dst,
                       int nl, int ns, int ks, int np, int kp)
{
    int i = blockIdx.x * blockDim.x + threadIdx.x;
    int total = nl * np * kp;
    if (i >= total) return;
    int k = i % kp;
    int n = (i / kp) % np;
    int l = i / (kp * np);
    float v = 0.f;
    if (n < ns && k < ks) v = src[((size_t)l * ns + n) * ks + k];
    dst[i] = v;
}

// ---------------------------------------------------------------------------
// Transpose+pad input: x (C=224, L) -> xp (L, 256) with channels 224..255 = 0
// ---------------------------------------------------------------------------
__global__ void xpose_k(const float* __restrict__ x, float* __restrict__ xp, int L)
{
    int i = blockIdx.x * blockDim.x + threadIdx.x;
    if (i >= L * DMODEL) return;
    int l = i >> 8, c = i & 255;
    xp[i] = (c < INCH) ? x[(size_t)c * L + l] : 0.f;
}

// ---------------------------------------------------------------------------
// LayerNorm over last dim (256). One wave32 per row, 8 elems/lane.
// ---------------------------------------------------------------------------
__global__ __launch_bounds__(256)
void layernorm_k(const float* __restrict__ s, const float* __restrict__ g,
                 const float* __restrict__ b, float* __restrict__ out, int L)
{
    int row  = (int)((blockIdx.x * blockDim.x + threadIdx.x) >> 5);
    int lane = threadIdx.x & 31;
    if (row >= L) return;
    const float* x = s + (size_t)row * DMODEL;
    float v[8]; float sum = 0.f;
    #pragma unroll
    for (int i = 0; i < 8; ++i) { v[i] = x[lane + 32 * i]; sum += v[i]; }
    #pragma unroll
    for (int o = 16; o; o >>= 1) sum += __shfl_xor(sum, o, 32);
    float mu = sum * (1.f / DMODEL);
    float var = 0.f;
    #pragma unroll
    for (int i = 0; i < 8; ++i) { float d = v[i] - mu; var += d * d; }
    #pragma unroll
    for (int o = 16; o; o >>= 1) var += __shfl_xor(var, o, 32);
    float rs = rsqrtf(var * (1.f / DMODEL) + 1e-5f);
    float* y = out + (size_t)row * DMODEL;
    #pragma unroll
    for (int i = 0; i < 8; ++i) {
        int c = lane + 32 * i;
        y[c] = (v[i] - mu) * rs * g[c] + b[c];
    }
}

// ---------------------------------------------------------------------------
// Causal depthwise conv (K=4) over u = xz[:, :512], fused SiLU.
// ---------------------------------------------------------------------------
__global__ void conv_silu_k(const float* __restrict__ xz, const float* __restrict__ cw,
                            const float* __restrict__ cb, float* __restrict__ uc, int total)
{
    int idx = blockIdx.x * blockDim.x + threadIdx.x;
    if (idx >= total) return;
    int l = idx >> 9, d = idx & 511;
    float a = cb[d];
    #pragma unroll
    for (int k = 0; k < KCONV; ++k) {
        int ls = l - (KCONV - 1) + k;
        if (ls >= 0) a += xz[(size_t)ls * (2 * DINNER) + d] * cw[d * KCONV + k];
    }
    uc[idx] = a * sigmoidf_(a);
}

// ---------------------------------------------------------------------------
// Selective scan: one thread per channel d, 16 states in VGPRs, serial over L.
// B/C rows (stride-64 padded x_dbl) are wave-uniform loads. Epilogue fuses the
// D skip-connection and the SiLU(z) gate -> yfin feeds out_proj directly.
// ---------------------------------------------------------------------------
__global__ __launch_bounds__(256)
void scan_k(const float* __restrict__ xz, const float* __restrict__ uc,
            const float* __restrict__ dt, const float* __restrict__ xdbl,
            const float* __restrict__ a_log, const float* __restrict__ dp,
            float* __restrict__ yfin, int L)
{
    int d = blockIdx.x * blockDim.x + threadIdx.x;
    if (d >= DINNER) return;
    float A[DSTATE];
    #pragma unroll
    for (int n = 0; n < DSTATE; ++n) A[n] = -__expf(a_log[d * DSTATE + n]);
    float Dd = dp[d];
    float h[DSTATE] = {};
    for (int l = 0; l < L; ++l) {
        float dtl = dt[(size_t)l * DINNER + d];
        float ul  = uc[(size_t)l * DINNER + d];
        float du  = dtl * ul;
        const float* bc = xdbl + (size_t)l * XDBP + DTRANK;   // B at +0, C at +16
        float y = 0.f;
        #pragma unroll
        for (int n = 0; n < DSTATE; ++n) {
            h[n] = h[n] * __expf(dtl * A[n]) + du * bc[n];
            y = fmaf(h[n], bc[DSTATE + n], y);
        }
        float z = xz[(size_t)l * (2 * DINNER) + DINNER + d];
        yfin[(size_t)l * DINNER + d] = (y + ul * Dd) * (z * sigmoidf_(z));
    }
}

__global__ void dup_k(const float* __restrict__ src, float* __restrict__ a,
                      float* __restrict__ b, int n)
{
    int i = blockIdx.x * blockDim.x + threadIdx.x;
    if (i < n) { float v = src[i]; a[i] = v; b[i] = v; }
}

__global__ void concat_k(const float* __restrict__ sa, const float* __restrict__ sb,
                         float* __restrict__ dst, int total)
{
    int i = blockIdx.x * blockDim.x + threadIdx.x;
    if (i >= total) return;
    int l = i >> 9, c = i & 511;
    dst[i] = (c < DMODEL) ? sa[(size_t)l * DMODEL + c]
                          : sb[(size_t)l * DMODEL + c - DMODEL];
}

// ---------------------------------------------------------------------------
extern "C" void kernel_launch(void* const* d_in, const int* in_sizes, int n_in,
                              void* d_out, int out_size, void* d_ws, size_t ws_size,
                              hipStream_t stream)
{
    (void)in_sizes; (void)n_in; (void)out_size; (void)ws_size;
    const float* x    = (const float*)d_in[0];
    const float* ipw  = (const float*)d_in[1];
    const float* ipb  = (const float*)d_in[2];
    const float* lng  = (const float*)d_in[3];
    const float* lnb  = (const float*)d_in[4];
    const float* inw  = (const float*)d_in[5];
    const float* cw   = (const float*)d_in[6];
    const float* cb   = (const float*)d_in[7];
    const float* xpw  = (const float*)d_in[8];
    const float* dtw  = (const float*)d_in[9];
    const float* dtbi = (const float*)d_in[10];
    const float* alog = (const float*)d_in[11];
    const float* dpar = (const float*)d_in[12];
    const float* ow   = (const float*)d_in[13];
    const float* fw1  = (const float*)d_in[14];
    const float* fb1  = (const float*)d_in[15];
    const float* fw2  = (const float*)d_in[16];
    const float* fb2  = (const float*)d_in[17];
    float* out = (float*)d_out;

    const int L = L_SEQ;
    float* p = (float*)d_ws;
    float* seq0   = p; p += (size_t)L * DMODEL;
    float* spec   = p; p += (size_t)L * DMODEL;
    float* spat   = p; p += (size_t)L * DMODEL;
    float* nrm    = p; p += (size_t)L * DMODEL;
    float* xz     = p; p += (size_t)L * 2 * DINNER;
    float* ucb    = p; p += (size_t)L * DINNER;
    float* xdb    = p; p += (size_t)L * XDBP;
    float* dtbuf  = p; p += (size_t)L * DINNER;
    float* yfin   = p; p += (size_t)L * DINNER;
    float* fused  = p; p += (size_t)L * 2 * DMODEL;
    float* fbuf   = p; p += (size_t)L * DMODEL;
    float* xpad   = p; p += (size_t)L * DMODEL;                       // transposed+padded input
    float* ipwp   = p; p += (size_t)DMODEL * DMODEL;                  // ipw padded K 224->256
    float* xpwp   = p; p += (size_t)NLAYERS * XDBP * DINNER;          // x_proj_w padded N 48->64
    float* dtwp   = p; p += (size_t)NLAYERS * DINNER * XDBP;          // dt_proj_w padded K 16->64

    dim3 blk(256);
    auto gg = [](int M, int N) { return dim3((unsigned)(N / 64), (unsigned)(M / 64)); };

    // ---- pre-pack: transpose/pad input, zero-pad ragged weights ----
    xpose_k<<<(L * DMODEL + 255) / 256, blk, 0, stream>>>(x, xpad, L);
    padw_k<<<(DMODEL * DMODEL + 255) / 256, blk, 0, stream>>>(
        ipw, ipwp, 1, DMODEL, INCH, DMODEL, DMODEL);
    padw_k<<<(NLAYERS * XDBP * DINNER + 255) / 256, blk, 0, stream>>>(
        xpw, xpwp, NLAYERS, DTRANK + 2 * DSTATE, DINNER, XDBP, DINNER);
    padw_k<<<(NLAYERS * DINNER * XDBP + 255) / 256, blk, 0, stream>>>(
        dtw, dtwp, NLAYERS, DINNER, DTRANK, DINNER, XDBP);

    // input projection: seq0 = xpad (L,256) x ipwp (256,256)^T + bias
    gemm_bf16_wmma<0, false><<<gg(L, DMODEL), blk, 0, stream>>>(
        xpad, ipwp, ipb, nullptr, seq0, L, DMODEL, DMODEL, DMODEL, DMODEL, DMODEL);
    dup_k<<<(L * DMODEL + 255) / 256, blk, 0, stream>>>(seq0, spec, spat, L * DMODEL);

    for (int i = 0; i < NLAYERS; ++i) {
        float* resid = (i < 4) ? spec : spat;
        layernorm_k<<<(L * 32 + 255) / 256, blk, 0, stream>>>(
            resid, lng + i * DMODEL, lnb + i * DMODEL, nrm, L);
        // in_proj: (L,256) x (1024,256)^T -> xz (L,1024)
        gemm_bf16_wmma<0, false><<<gg(L, 2 * DINNER), blk, 0, stream>>>(
            nrm, inw + (size_t)i * 2 * DINNER * DMODEL, nullptr, nullptr, xz,
            L, 2 * DINNER, DMODEL, DMODEL, DMODEL, 2 * DINNER);
        // causal depthwise conv + SiLU on u = xz[:, :512]
        conv_silu_k<<<(L * DINNER + 255) / 256, blk, 0, stream>>>(
            xz, cw + (size_t)i * DINNER * KCONV, cb + (size_t)i * DINNER, ucb, L * DINNER);
        // x_proj: (L,512) x (64,512)^T -> xdb (L,64)  (rows 48..63 zero)
        gemm_bf16_wmma<0, false><<<gg(L, XDBP), blk, 0, stream>>>(
            ucb, xpwp + (size_t)i * XDBP * DINNER, nullptr, nullptr, xdb,
            L, XDBP, DINNER, DINNER, DINNER, XDBP);
        // dt_proj + bias + softplus: (L,64) x (512,64)^T -> dtbuf (L,512)
        // (A cols 16..63 hold B/C values but the padded weight cols are zero)
        gemm_bf16_wmma<2, false><<<gg(L, DINNER), blk, 0, stream>>>(
            xdb, dtwp + (size_t)i * DINNER * XDBP, dtbi + (size_t)i * DINNER, nullptr, dtbuf,
            L, DINNER, XDBP, XDBP, XDBP, DINNER);
        // selective scan (serial in L, parallel over 512 channels) + D skip + SiLU(z) gate
        scan_k<<<(DINNER + 255) / 256, blk, 0, stream>>>(
            xz, ucb, dtbuf, xdb, alog + (size_t)i * DINNER * DSTATE,
            dpar + (size_t)i * DINNER, yfin, L);
        // out_proj with fused residual: resid += (L,512) x (256,512)^T
        gemm_bf16_wmma<0, false><<<gg(L, DMODEL), blk, 0, stream>>>(
            yfin, ow + (size_t)i * DMODEL * DINNER, nullptr, resid, resid,
            L, DMODEL, DINNER, DINNER, DINNER, DMODEL);
    }

    // fusion head
    concat_k<<<(L * 2 * DMODEL + 255) / 256, blk, 0, stream>>>(spec, spat, fused, L * 2 * DMODEL);
    gemm_bf16_wmma<1, false><<<gg(L, DMODEL), blk, 0, stream>>>(
        fused, fw1, fb1, nullptr, fbuf, L, DMODEL, 2 * DMODEL, 2 * DMODEL, 2 * DMODEL, DMODEL);
    // final projection, stored transposed -> (LAT, L) == (B, LAT, H, W)
    gemm_bf16_wmma<0, true><<<gg(L, LATD), blk, 0, stream>>>(
        fbuf, fw2, fb2, nullptr, out, L, LATD, DMODEL, DMODEL, DMODEL, L);
}
